// SwinTransformerBlock_16097537426159
// MI455X (gfx1250) — matrix operand order repaired and checked by reading, not compiled, not run on hardware
//
#include <hip/hip_runtime.h>
#include <hip/hip_bf16.h>

typedef __attribute__((ext_vector_type(16))) _Float16 v16h;
typedef __attribute__((ext_vector_type(8)))  float    v8f;
typedef __attribute__((ext_vector_type(4)))  _Float16 v4h;

#define WS7   7
#define NH4   4
#define HD32  32
#define C128  128
#define HID512 512
#define SHIFT3 3
#define IMG56 56
#define NWIN  4096          // 64 batches * 8*8 windows
#define WT49  49
#define MTOT  200704        // NWIN*49 == B*L

// ---------------------------------------------------------------------------
// CDNA5 async copy: global -> LDS, 16B per active lane, tracked by ASYNCcnt.
// (ISA 10. "Global Async LDS": GLOBAL_LOAD_ASYNC_TO_LDS_B128, GV mode)
// lds_addr: wave-relative LDS byte address (low 32 bits of generic pointer).
// ---------------------------------------------------------------------------
__device__ __forceinline__ void async_copy_b128(unsigned lds_addr, const void* gptr) {
    asm volatile("global_load_async_to_lds_b128 %0, %1, off"
                 :: "v"(lds_addr), "v"((unsigned long long)(uintptr_t)gptr)
                 : "memory");
}
__device__ __forceinline__ void wait_async0() {
    asm volatile("s_wait_asynccnt 0x0" ::: "memory");
}

// ---------------------------------------------------------------------------
// CDNA5 WMMA fragment loaders (wave32). LDS tiles are row-major, K contiguous.
// A 16x32 f16: lane r=lane&15, koff=(lane>=16)*8; halves 0..7 -> K=koff..koff+7,
// halves 8..15 -> K=16+koff..  (ISA 7.12.2 "16-bit A-Matrix 16x32")
// ---------------------------------------------------------------------------
__device__ __forceinline__ v16h load_frag_a(const _Float16* base, int ld, int k0) {
    const int lane = threadIdx.x & 31;
    const int r    = lane & 15;
    const int koff = (lane >> 4) << 3;
    const _Float16* p = base + r * ld + k0 + koff;
    v16h a;
#pragma unroll
    for (int h = 0; h < 8; ++h) a[h] = p[h];
#pragma unroll
    for (int h = 0; h < 8; ++h) a[8 + h] = p[16 + h];
    return a;
}

// B 32x16 f16 fed from B^T stored [n][K] (K contiguous): lane n=lane&15,
// kbase=(lane>=16)*8; halves 0..7 -> K=kbase+h, 8..15 -> K=16+kbase+(h-8).
__device__ __forceinline__ v16h load_frag_b(const _Float16* base, int ld, int k0) {
    const int lane  = threadIdx.x & 31;
    const int n     = lane & 15;
    const int kbase = (lane >> 4) << 3;
    const _Float16* p = base + n * ld + k0 + kbase;
    v16h b;
#pragma unroll
    for (int h = 0; h < 16; ++h) b[h] = p[h < 8 ? h : h + 8];
    return b;
}

__device__ __forceinline__ v8f wmma_f16(v16h a, v16h b, v8f c) {
    return __builtin_amdgcn_wmma_f32_16x16x32_f16(false, a, false, b,
                                                  (short)0, c, false, false);
}

// ---------------------------------------------------------------------------
// Kernel 1: LN1 + cyclic shift + window partition, f32 -> f16.
// One wave per token (C=128 -> 4 floats/lane), 8 tokens per 256-thr block.
// ---------------------------------------------------------------------------
__global__ void __launch_bounds__(256)
k_ln1_window(const float* __restrict__ x, const float* __restrict__ g,
             const float* __restrict__ bt, _Float16* __restrict__ hwin)
{
    const int t    = blockIdx.x * 8 + (threadIdx.x >> 5);
    const int lane = threadIdx.x & 31;
    const int w    = t / WT49;
    const int idx  = t % WT49;
    const int b    = w >> 6;
    const int wrow = (w >> 3) & 7;
    const int wcol = w & 7;
    const int r    = idx / WS7;
    const int c2   = idx % WS7;
    const int hs   = (wrow * WS7 + r  + SHIFT3) % IMG56;   // roll(-SHIFT)
    const int wsp  = (wcol * WS7 + c2 + SHIFT3) % IMG56;

    const float4 v = ((const float4*)(x + ((size_t)b * 3136 + hs * IMG56 + wsp) * C128))[lane];
    float s  = v.x + v.y + v.z + v.w;
    float s2 = v.x * v.x + v.y * v.y + v.z * v.z + v.w * v.w;
#pragma unroll
    for (int d = 16; d; d >>= 1) { s += __shfl_xor(s, d); s2 += __shfl_xor(s2, d); }
    const float mu  = s * (1.0f / 128.0f);
    const float var = s2 * (1.0f / 128.0f) - mu * mu;
    const float inv = rsqrtf(var + 1e-5f);

    const float4 gg = ((const float4*)g)[lane];
    const float4 bb = ((const float4*)bt)[lane];
    v4h o;
    o[0] = (_Float16)((v.x - mu) * inv * gg.x + bb.x);
    o[1] = (_Float16)((v.y - mu) * inv * gg.y + bb.y);
    o[2] = (_Float16)((v.z - mu) * inv * gg.z + bb.z);
    o[3] = (_Float16)((v.w - mu) * inv * gg.w + bb.w);
    ((v4h*)(hwin + (size_t)t * C128))[lane] = o;
}

// ---------------------------------------------------------------------------
// Kernel 2: QKV GEMM  (M=200704, K=128, N=384) out = hwin @ qkv_w^T + qkv_b.
// A tile staged with global_load_async_to_lds_b128; weights converted via VGPRs.
// Scatter q/k/v into (win, head, 49, 32) f16; q pre-scaled by head_dim^-0.5.
// ---------------------------------------------------------------------------
__global__ void __launch_bounds__(256)
k_gemm_qkv(const _Float16* __restrict__ hwin, const float* __restrict__ qkv_w,
           const float* __restrict__ qkv_b, _Float16* __restrict__ qbuf,
           _Float16* __restrict__ kbuf, _Float16* __restrict__ vbuf)
{
    __shared__ _Float16 sA[64 * 128];    // activations (K contiguous)
    __shared__ _Float16 sB[128 * 128];   // B^T = qkv_w rows (K contiguous)
    const int m0  = blockIdx.x * 64;
    const int n0  = blockIdx.y * 128;
    const int tid = threadIdx.x;

    {   // stage A: 1024 x 16B async global->LDS (no VGPR round-trip)
        const unsigned ldsA = (unsigned)(uintptr_t)sA;
        const uint4* src = (const uint4*)(hwin + (size_t)m0 * C128);
#pragma unroll
        for (int i = 0; i < 4; ++i)
            async_copy_b128(ldsA + (unsigned)(tid + 256 * i) * 16u, src + tid + 256 * i);
    }
    {   // stage B (f32 -> f16): 4096 quads
        const float4* src = (const float4*)(qkv_w + (size_t)n0 * C128);
#pragma unroll
        for (int i = 0; i < 16; ++i) {
            float4 f = src[tid + 256 * i];
            v4h h4; h4[0] = (_Float16)f.x; h4[1] = (_Float16)f.y;
            h4[2] = (_Float16)f.z; h4[3] = (_Float16)f.w;
            ((v4h*)sB)[tid + 256 * i] = h4;
        }
    }
    wait_async0();
    __syncthreads();

    const int wid = tid >> 5, lane = tid & 31;
    const int mt  = wid >> 1;
    const int ntb = (wid & 1) * 4;
    v8f acc[4] = {};
#pragma unroll
    for (int ks = 0; ks < 4; ++ks) {
        v16h a = load_frag_a(sA + mt * 16 * 128, 128, ks * 32);
#pragma unroll
        for (int j = 0; j < 4; ++j) {
            v16h b = load_frag_b(sB + (ntb + j) * 16 * 128, 128, ks * 32);
            acc[j] = wmma_f16(a, b, acc[j]);
        }
    }

    const int colL  = lane & 15;
    const int rbase = (lane >> 4) * 8;
    const float qscale = 0.17677669529663687f;    // 32^-0.5
    const int which = n0 >> 7;                    // 0=q 1=k 2=v (uniform)
    _Float16* dstbase = which == 0 ? qbuf : (which == 1 ? kbuf : vbuf);
#pragma unroll
    for (int j = 0; j < 4; ++j) {
        const int n    = n0 + (ntb + j) * 16 + colL;
        const float bias = qkv_b[n];
        const int cn   = n & 127;
        const int head = cn >> 5, dch = cn & 31;
#pragma unroll
        for (int g = 0; g < 8; ++g) {
            const int m = m0 + mt * 16 + rbase + g;
            const int w = m / WT49, idx = m % WT49;
            float v = acc[j][g] + bias;
            if (which == 0) v *= qscale;
            dstbase[(((size_t)w * NH4 + head) * WT49 + idx) * HD32 + dch] = (_Float16)v;
        }
    }
}

// ---------------------------------------------------------------------------
// Kernel 3: windowed attention, one 4-wave block per (window, head).
// Q/K staged via async copy (196 x 16B) + zero-fill of pad rows; V transposed
// during staging. S = QK^T + rel-pos bias, in-register softmax via shfl over
// the 16-lane halves of the C fragment, then O = A @ V via WMMA.
// ---------------------------------------------------------------------------
__global__ void __launch_bounds__(128)
k_attn(const _Float16* __restrict__ qbuf, const _Float16* __restrict__ kbuf,
       const _Float16* __restrict__ vbuf, const float* __restrict__ rpb,
       _Float16* __restrict__ obuf)
{
    __shared__ _Float16 sQ[64 * 32];   // A layout (K=32 contiguous), rows>=49 zero
    __shared__ _Float16 sK[64 * 32];   // B^T for QK^T == k row-major
    __shared__ _Float16 sA[64 * 64];   // softmaxed probs, A layout
    __shared__ _Float16 sV[32 * 64];   // B^T for A@V == v transposed
    const int wh   = blockIdx.x;
    const int head = wh & 3;
    const int win  = wh >> 2;
    const int tid  = threadIdx.x;

    const _Float16* qsrc = qbuf + (size_t)wh * WT49 * HD32;
    const _Float16* ksrc = kbuf + (size_t)wh * WT49 * HD32;
    const _Float16* vsrc = vbuf + (size_t)wh * WT49 * HD32;
    {   // async stage Q,K: 49*32*2B = 3136B = 196 x 16B chunks each
        const unsigned ldsQ = (unsigned)(uintptr_t)sQ;
        const unsigned ldsK = (unsigned)(uintptr_t)sK;
        for (int c = tid; c < 196; c += 128) {
            async_copy_b128(ldsQ + (unsigned)c * 16u, (const uint4*)qsrc + c);
            async_copy_b128(ldsK + (unsigned)c * 16u, (const uint4*)ksrc + c);
        }
        for (int i = WT49 * 32 + tid; i < 64 * 32; i += 128) {  // zero pad rows
            sQ[i] = (_Float16)0.0f;
            sK[i] = (_Float16)0.0f;
        }
    }
    for (int i = tid; i < 32 * 64; i += 128) {  // V^T with K padding
        const int n = i >> 6, kk = i & 63;
        sV[i] = kk < WT49 ? vsrc[kk * HD32 + n] : (_Float16)0.0f;
    }
    wait_async0();
    __syncthreads();

    const int wv = tid >> 5, lane = tid & 31;
    const int colL = lane & 15, rbase = (lane >> 4) * 8;

    v8f s[4] = {};
    {
        v16h a = load_frag_a(sQ + wv * 16 * 32, 32, 0);
#pragma unroll
        for (int j = 0; j < 4; ++j) {
            v16h b = load_frag_b(sK + j * 16 * 32, 32, 0);
            s[j] = wmma_f16(a, b, s[j]);
        }
    }

    // bias + mask + softmax per row (row lives on the 16 lanes of a half-wave)
#pragma unroll
    for (int g = 0; g < 8; ++g) {
        const int row = wv * 16 + rbase + g;
        const int yi = row / WS7, xi = row % WS7;
#pragma unroll
        for (int j = 0; j < 4; ++j) {
            const int col = j * 16 + colL;
            float v = s[j][g];
            if (row < WT49 && col < WT49) {
                const int yj = col / WS7, xj = col % WS7;
                const int ridx = (yi - yj + WS7 - 1) * (2 * WS7 - 1) + (xi - xj + WS7 - 1);
                v += rpb[ridx * NH4 + head];
            }
            if (col >= WT49) v = -1e30f;
            s[j][g] = v;
        }
        float mx = fmaxf(fmaxf(s[0][g], s[1][g]), fmaxf(s[2][g], s[3][g]));
#pragma unroll
        for (int d = 8; d; d >>= 1) mx = fmaxf(mx, __shfl_xor(mx, d));
        float sm = 0.0f;
#pragma unroll
        for (int j = 0; j < 4; ++j) { float e = __expf(s[j][g] - mx); s[j][g] = e; sm += e; }
#pragma unroll
        for (int d = 8; d; d >>= 1) sm += __shfl_xor(sm, d);
        const float rinv = 1.0f / sm;
#pragma unroll
        for (int j = 0; j < 4; ++j)
            sA[row * 64 + j * 16 + colL] = (_Float16)(s[j][g] * rinv);
    }
    __syncthreads();

    v8f o[2] = {};
#pragma unroll
    for (int ks = 0; ks < 2; ++ks) {
        v16h a = load_frag_a(sA + wv * 16 * 64, 64, ks * 32);
#pragma unroll
        for (int j = 0; j < 2; ++j) {
            v16h b = load_frag_b(sV + j * 16 * 64, 64, ks * 32);
            o[j] = wmma_f16(a, b, o[j]);
        }
    }
#pragma unroll
    for (int j = 0; j < 2; ++j) {
        const int cc = head * HD32 + j * 16 + colL;
#pragma unroll
        for (int g = 0; g < 8; ++g) {
            const int row = wv * 16 + rbase + g;
            if (row < WT49)
                obuf[((size_t)win * WT49 + row) * C128 + cc] = (_Float16)o[j][g];
        }
    }
}

// ---------------------------------------------------------------------------
// Kernel 4: proj GEMM (M=200704, K=128, N=128) + window reverse + unshift +
// residual add with original x  ->  xmid (f32).
// ---------------------------------------------------------------------------
__global__ void __launch_bounds__(256)
k_gemm_proj(const _Float16* __restrict__ obuf, const float* __restrict__ proj_w,
            const float* __restrict__ proj_b, const float* __restrict__ x,
            float* __restrict__ xmid)
{
    __shared__ _Float16 sA[64 * 128];
    __shared__ _Float16 sB[128 * 128];
    const int m0  = blockIdx.x * 64;
    const int tid = threadIdx.x;
    {   // async stage A
        const unsigned ldsA = (unsigned)(uintptr_t)sA;
        const uint4* src = (const uint4*)(obuf + (size_t)m0 * C128);
#pragma unroll
        for (int i = 0; i < 4; ++i)
            async_copy_b128(ldsA + (unsigned)(tid + 256 * i) * 16u, src + tid + 256 * i);
    }
    {
        const float4* src = (const float4*)proj_w;
#pragma unroll
        for (int i = 0; i < 16; ++i) {
            float4 f = src[tid + 256 * i];
            v4h h4; h4[0] = (_Float16)f.x; h4[1] = (_Float16)f.y;
            h4[2] = (_Float16)f.z; h4[3] = (_Float16)f.w;
            ((v4h*)sB)[tid + 256 * i] = h4;
        }
    }
    wait_async0();
    __syncthreads();

    const int wid = tid >> 5, lane = tid & 31;
    const int mt  = wid >> 1;
    const int ntb = (wid & 1) * 4;
    v8f acc[4] = {};
#pragma unroll
    for (int ks = 0; ks < 4; ++ks) {
        v16h a = load_frag_a(sA + mt * 16 * 128, 128, ks * 32);
#pragma unroll
        for (int j = 0; j < 4; ++j) {
            v16h b = load_frag_b(sB + (ntb + j) * 16 * 128, 128, ks * 32);
            acc[j] = wmma_f16(a, b, acc[j]);
        }
    }

    const int colL = lane & 15, rbase = (lane >> 4) * 8;
#pragma unroll
    for (int j = 0; j < 4; ++j) {
        const int n = (ntb + j) * 16 + colL;
        const float bias = proj_b[n];
#pragma unroll
        for (int g = 0; g < 8; ++g) {
            const int m = m0 + mt * 16 + rbase + g;
            const int w = m / WT49, idx = m % WT49;
            const int b = w >> 6, wrow = (w >> 3) & 7, wcol = w & 7;
            const int r = idx / WS7, c2 = idx % WS7;
            const int hh = (wrow * WS7 + r  + SHIFT3) % IMG56;   // roll(+SHIFT)
            const int ww = (wcol * WS7 + c2 + SHIFT3) % IMG56;
            const size_t di = ((size_t)b * 3136 + hh * IMG56 + ww) * C128 + n;
            xmid[di] = x[di] + acc[j][g] + bias;
        }
    }
}

// ---------------------------------------------------------------------------
// Kernel 5: fused MLP. Per 64-row block: LN2 -> LDS f16, FC1(128->512) in 4
// N-chunks with GELU into 64KB LDS, FC2(512->128) in 4 K-chunks accumulating
// in registers, + residual. Hidden tensor (205 MB) never reaches HBM.
// ---------------------------------------------------------------------------
__global__ void __launch_bounds__(256)
k_mlp(const float* __restrict__ xmid, const float* __restrict__ g2,
      const float* __restrict__ b2, const float* __restrict__ fc1_w,
      const float* __restrict__ fc1_b, const float* __restrict__ fc2_w,
      const float* __restrict__ fc2_b, float* __restrict__ out)
{
    __shared__ _Float16 sX[64 * 128];     // 16KB LN'd activations
    __shared__ _Float16 sH[64 * 512];     // 64KB GELU(FC1) hidden
    __shared__ _Float16 sW[128 * 128];    // 32KB weight staging
    const int m0  = blockIdx.x * 64;
    const int tid = threadIdx.x;
    const int wid = tid >> 5, lane = tid & 31;

    // Phase 1: LN2 (8 waves x 8 rows)
#pragma unroll
    for (int it = 0; it < 8; ++it) {
        const int row = wid * 8 + it;
        const float4 v = ((const float4*)(xmid + ((size_t)m0 + row) * C128))[lane];
        float s  = v.x + v.y + v.z + v.w;
        float s2 = v.x * v.x + v.y * v.y + v.z * v.z + v.w * v.w;
#pragma unroll
        for (int d = 16; d; d >>= 1) { s += __shfl_xor(s, d); s2 += __shfl_xor(s2, d); }
        const float mu  = s * (1.0f / 128.0f);
        const float var = s2 * (1.0f / 128.0f) - mu * mu;
        const float inv = rsqrtf(var + 1e-5f);
        const float4 gg = ((const float4*)g2)[lane];
        const float4 bb = ((const float4*)b2)[lane];
        v4h o;
        o[0] = (_Float16)((v.x - mu) * inv * gg.x + bb.x);
        o[1] = (_Float16)((v.y - mu) * inv * gg.y + bb.y);
        o[2] = (_Float16)((v.z - mu) * inv * gg.z + bb.z);
        o[3] = (_Float16)((v.w - mu) * inv * gg.w + bb.w);
        ((v4h*)(sX + row * 128))[lane] = o;
    }
    __syncthreads();

    const int mt  = wid >> 1;
    const int ntb = (wid & 1) * 4;
    const int colL = lane & 15, rbase = (lane >> 4) * 8;

    // Phase 2: FC1 + GELU, N=512 in 4 chunks of 128
    for (int nc = 0; nc < 4; ++nc) {
        {
            const float4* src = (const float4*)(fc1_w + (size_t)nc * 128 * 128);
#pragma unroll
            for (int i = 0; i < 16; ++i) {
                float4 f = src[tid + 256 * i];
                v4h h4; h4[0] = (_Float16)f.x; h4[1] = (_Float16)f.y;
                h4[2] = (_Float16)f.z; h4[3] = (_Float16)f.w;
                ((v4h*)sW)[tid + 256 * i] = h4;
            }
        }
        __syncthreads();
        v8f acc[4] = {};
#pragma unroll
        for (int ks = 0; ks < 4; ++ks) {
            v16h a = load_frag_a(sX + mt * 16 * 128, 128, ks * 32);
#pragma unroll
            for (int j = 0; j < 4; ++j) {
                v16h b = load_frag_b(sW + (ntb + j) * 16 * 128, 128, ks * 32);
                acc[j] = wmma_f16(a, b, acc[j]);
            }
        }
#pragma unroll
        for (int j = 0; j < 4; ++j) {
            const int n = nc * 128 + (ntb + j) * 16 + colL;
            const float bias = fc1_b[n];
#pragma unroll
            for (int g = 0; g < 8; ++g) {
                float vv = acc[j][g] + bias;
                vv = 0.5f * vv * (1.0f + erff(vv * 0.70710678118654752f));  // exact GELU
                sH[(mt * 16 + rbase + g) * 512 + n] = (_Float16)vv;
            }
        }
        __syncthreads();
    }

    // Phase 3: FC2, K=512 in 4 chunks; accumulate in registers
    v8f acc[4] = {};
    for (int kc = 0; kc < 4; ++kc) {
        {
            const int base = kc * 128;
#pragma unroll
            for (int i = 0; i < 16; ++i) {
                const int e = tid + 256 * i;      // quad index: 128 rows x 32 quads
                const int n = e >> 5, q = e & 31;
                float4 f = ((const float4*)(fc2_w + (size_t)n * 512 + base))[q];
                v4h h4; h4[0] = (_Float16)f.x; h4[1] = (_Float16)f.y;
                h4[2] = (_Float16)f.z; h4[3] = (_Float16)f.w;
                ((v4h*)(sW + n * 128))[q] = h4;
            }
        }
        __syncthreads();
#pragma unroll
        for (int ks = 0; ks < 4; ++ks) {
            v16h a = load_frag_a(sH + mt * 16 * 512, 512, kc * 128 + ks * 32);
#pragma unroll
            for (int j = 0; j < 4; ++j) {
                v16h b = load_frag_b(sW + (ntb + j) * 16 * 128, 128, ks * 32);
                acc[j] = wmma_f16(a, b, acc[j]);
            }
        }
        __syncthreads();
    }

#pragma unroll
    for (int j = 0; j < 4; ++j) {
        const int n = (ntb + j) * 16 + colL;
        const float bias = fc2_b[n];
#pragma unroll
        for (int g = 0; g < 8; ++g) {
            const size_t m = (size_t)m0 + mt * 16 + rbase + g;
            out[m * C128 + n] = xmid[m * C128 + n] + acc[j][g] + bias;
        }
    }
}

// ---------------------------------------------------------------------------
extern "C" void kernel_launch(void* const* d_in, const int* in_sizes, int n_in,
                              void* d_out, int out_size, void* d_ws, size_t ws_size,
                              hipStream_t stream)
{
    (void)in_sizes; (void)n_in; (void)out_size; (void)ws_size;
    const float* x      = (const float*)d_in[0];
    const float* n1g    = (const float*)d_in[1];
    const float* n1b    = (const float*)d_in[2];
    const float* qkv_w  = (const float*)d_in[3];
    const float* qkv_b  = (const float*)d_in[4];
    const float* proj_w = (const float*)d_in[5];
    const float* proj_b = (const float*)d_in[6];
    const float* rpb    = (const float*)d_in[7];
    const float* n2g    = (const float*)d_in[8];
    const float* n2b    = (const float*)d_in[9];
    const float* fc1_w  = (const float*)d_in[10];
    const float* fc1_b  = (const float*)d_in[11];
    const float* fc2_w  = (const float*)d_in[12];
    const float* fc2_b  = (const float*)d_in[13];
    float* out = (float*)d_out;

    char* ws = (char*)d_ws;
    const size_t M = MTOT;
    _Float16* hwin = (_Float16*)ws;  ws += M * C128 * sizeof(_Float16);
    _Float16* qbuf = (_Float16*)ws;  ws += M * C128 * sizeof(_Float16);
    _Float16* kbuf = (_Float16*)ws;  ws += M * C128 * sizeof(_Float16);
    _Float16* vbuf = (_Float16*)ws;  ws += M * C128 * sizeof(_Float16);
    _Float16* obuf = (_Float16*)ws;  ws += M * C128 * sizeof(_Float16);
    float*    xmid = (float*)ws;     ws += M * C128 * sizeof(float);

    k_ln1_window<<<M / 8, 256, 0, stream>>>(x, n1g, n1b, hwin);
    k_gemm_qkv<<<dim3(M / 64, 3), 256, 0, stream>>>(hwin, qkv_w, qkv_b, qbuf, kbuf, vbuf);
    k_attn<<<NWIN * NH4, 128, 0, stream>>>(qbuf, kbuf, vbuf, rpb, obuf);
    k_gemm_proj<<<M / 64, 256, 0, stream>>>(obuf, proj_w, proj_b, x, xmid);
    k_mlp<<<M / 64, 256, 0, stream>>>(xmid, n2g, n2b, fc1_w, fc1_b, fc2_w, fc2_b, out);
}
